// LEAM_17205638988540
// MI455X (gfx1250) — compile-verified, matrix operand order checked by reading, not run on hardware
//
#include <hip/hip_runtime.h>
#include <hip/hip_bf16.h>

// LEAM forward: embed -> cosine sim (WMMA f16) -> conv/relu/max -> softmax
// -> attention pool -> linear head.
// B=32, L=2048, P=512, K=64, VOCAB=50257, window=11 (R=5)

#define LEAM_B 32
#define LEAM_L 2048
#define LEAM_P 512
#define LEAM_K 64
#define ASTR   520   // padded f16 row stride: 1040B = 65*16B aligned, bank-shift 4/row

typedef __attribute__((ext_vector_type(16))) _Float16 v16h;
typedef __attribute__((ext_vector_type(8)))  float    v8f;

// ---------------------------------------------------------------- kernel 0
// Normalize label embeddings C[64,512] -> f16 Cn[64,512]; one wave per row.
__global__ void __launch_bounds__(32)
leam_cnorm_kernel(const float* __restrict__ C, _Float16* __restrict__ Cn) {
    const int k    = blockIdx.x;
    const int lane = threadIdx.x;
    const float* row = C + k * LEAM_P;
    float v[16];
    float ss = 0.f;
#pragma unroll
    for (int i = 0; i < 16; ++i) { float t = row[i * 32 + lane]; v[i] = t; ss += t * t; }
#pragma unroll
    for (int m = 16; m >= 1; m >>= 1) ss += __shfl_xor(ss, m, 32);
    const float rn = rsqrtf(ss);
#pragma unroll
    for (int i = 0; i < 16; ++i) Cn[k * LEAM_P + i * 32 + lane] = (_Float16)(v[i] * rn);
}

// ---------------------------------------------------------------- kernel 1
// Cosine-sim GEMM: G[b,k,l] = l_hat[k,:] . x_hat[b,l,:]
// Block = 64 threads (2 waves) handles one (b, 64-row L tile).
// Each wave: 32 L-rows x 64 K cols = 8 WMMA acc chains; per K-step:
// 2 A-fragments + 4 B-fragments (ds_load_b128 pairs) feed 8 wmmas.
__global__ void __launch_bounds__(64)
leam_sim_kernel(const int* __restrict__ x, const float* __restrict__ V,
                const _Float16* __restrict__ Cn, float* __restrict__ G) {
    extern __shared__ _Float16 sm[];
    _Float16* Ash = sm;                 // [64][ASTR] normalized emb tile (f16)
    _Float16* Bsh = sm + 64 * ASTR;     // [64][ASTR] normalized labels (f16)

    const int tid  = threadIdx.x;
    const int wave = tid >> 5;          // 0..1
    const int lane = tid & 31;
    const int b    = blockIdx.x >> 5;          // 32 tiles per batch
    const int lt   = (blockIdx.x & 31) << 6;   // tile base in L

    // Stage Cn into LDS (16B chunks).
    for (int i = tid; i < 64 * (LEAM_P / 8); i += 64) {
        const int r = i >> 6, c = (i & 63) << 3;
        *(uint4*)(Bsh + r * ASTR + c) = *(const uint4*)(Cn + r * LEAM_P + c);
    }
    // Gather + L2-normalize 64 embedding rows; one wave per row, 32 rows/wave.
    for (int r = wave; r < 64; r += 2) {
        const int tok = x[b * LEAM_L + lt + r];
        const float* row = V + (long long)tok * LEAM_P;
        float v[16];
        float ss = 0.f;
#pragma unroll
        for (int i = 0; i < 16; ++i) { float t = row[i * 32 + lane]; v[i] = t; ss += t * t; }
#pragma unroll
        for (int m = 16; m >= 1; m >>= 1) ss += __shfl_xor(ss, m, 32);
        const float rn = rsqrtf(ss);
#pragma unroll
        for (int i = 0; i < 16; ++i)
            Ash[r * ASTR + i * 32 + lane] = (_Float16)(v[i] * rn);
    }
    __syncthreads();

    // 16-bit WMMA lane layout: lane group g holds K in
    // {8g..8g+7} U {16+8g..23+8g} of each 32-chunk -> two 16B runs.
    const int lrow = lane & 15;
    const int grp  = lane >> 4;
    const int kb   = grp << 3;
    const int m0   = wave << 5;         // 32 rows per wave

    union Frag { v16h v; uint4 q[2]; };
    v8f acc[8];
#pragma unroll
    for (int t = 0; t < 8; ++t) acc[t] = (v8f)0.f;

    for (int p0 = 0; p0 < LEAM_P; p0 += 32) {
        const _Float16* ar0 = Ash + (m0 + lrow) * ASTR + p0 + kb;
        const _Float16* ar1 = ar0 + 16 * ASTR;
        Frag a0, a1;
        a0.q[0] = *(const uint4*)(ar0);
        a0.q[1] = *(const uint4*)(ar0 + 16);
        a1.q[0] = *(const uint4*)(ar1);
        a1.q[1] = *(const uint4*)(ar1 + 16);
#pragma unroll
        for (int nt = 0; nt < 4; ++nt) {
            const _Float16* br = Bsh + (nt * 16 + lrow) * ASTR + p0 + kb;
            Frag bf;
            bf.q[0] = *(const uint4*)(br);
            bf.q[1] = *(const uint4*)(br + 16);
            acc[nt] = __builtin_amdgcn_wmma_f32_16x16x32_f16(
                false, a0.v, false, bf.v, (short)0, acc[nt], false, false);
            acc[nt + 4] = __builtin_amdgcn_wmma_f32_16x16x32_f16(
                false, a1.v, false, bf.v, (short)0, acc[nt + 4], false, false);
        }
    }

    // D layout: lane group g owns rows M = 8g..8g+7 (VGPR r), col N = lane&15.
#pragma unroll
    for (int mt = 0; mt < 2; ++mt) {
#pragma unroll
        for (int nt = 0; nt < 4; ++nt) {
            const int kcol = nt * 16 + lrow;
            const v8f a = acc[mt * 4 + nt];
#pragma unroll
            for (int r = 0; r < 8; ++r) {
                const int l = lt + m0 + mt * 16 + (grp << 3) + r;
                G[((b << 6) + kcol) * LEAM_L + l] = a[r];
            }
        }
    }
}

// ---------------------------------------------------------------- kernel 2
// Per batch: 11-tap conv over L, +bias, relu, max over K, softmax over L.
__global__ void __launch_bounds__(256)
leam_score_kernel(const float* __restrict__ G, const float* __restrict__ f1_w,
                  const float* __restrict__ f1_b, float* __restrict__ beta) {
    __shared__ float srow[LEAM_L];
    __shared__ float sbest[LEAM_L];
    __shared__ float red[8];
    __shared__ float wsh[11];
    __shared__ float bsh[LEAM_K];
    const int b = blockIdx.x, tid = threadIdx.x;
    if (tid < 11) wsh[tid] = f1_w[tid];
    if (tid < LEAM_K) bsh[tid] = f1_b[tid];
    for (int l = tid; l < LEAM_L; l += 256) sbest[l] = 0.f;  // relu => max >= 0
    __syncthreads();

    for (int k = 0; k < LEAM_K; ++k) {
        const float* g = G + (b * LEAM_K + k) * LEAM_L;
        for (int l = tid; l < LEAM_L; l += 256) srow[l] = g[l];
        __syncthreads();
        const float fb = bsh[k];
        for (int l = tid; l < LEAM_L; l += 256) {
            float acc = 0.f;
#pragma unroll
            for (int d = 0; d < 11; ++d) {
                const int ll = l + d - 5;
                const float gv = (ll >= 0 && ll < LEAM_L) ? srow[ll] : 0.f;
                acc += wsh[d] * gv;
            }
            const float m = fmaxf(acc + fb, 0.f);
            sbest[l] = fmaxf(sbest[l], m);
        }
        __syncthreads();
    }

    // softmax over L
    float lmax = -3.4e38f;
    for (int l = tid; l < LEAM_L; l += 256) lmax = fmaxf(lmax, sbest[l]);
#pragma unroll
    for (int m = 16; m >= 1; m >>= 1) lmax = fmaxf(lmax, __shfl_xor(lmax, m, 32));
    if ((tid & 31) == 0) red[tid >> 5] = lmax;
    __syncthreads();
    const float gmax = fmaxf(fmaxf(fmaxf(red[0], red[1]), fmaxf(red[2], red[3])),
                             fmaxf(fmaxf(red[4], red[5]), fmaxf(red[6], red[7])));
    float lsum = 0.f;
    for (int l = tid; l < LEAM_L; l += 256) lsum += __expf(sbest[l] - gmax);
#pragma unroll
    for (int m = 16; m >= 1; m >>= 1) lsum += __shfl_xor(lsum, m, 32);
    __syncthreads();
    if ((tid & 31) == 0) red[tid >> 5] = lsum;
    __syncthreads();
    const float inv = 1.f / (red[0] + red[1] + red[2] + red[3] +
                             red[4] + red[5] + red[6] + red[7]);
    for (int l = tid; l < LEAM_L; l += 256)
        beta[b * LEAM_L + l] = __expf(sbest[l] - gmax) * inv;
}

// ---------------------------------------------------------------- kernel 3
// pooled[b,p] = mean_l beta[b,l] * V[x[b,l], p]   (raw embeddings)
__global__ void __launch_bounds__(256)
leam_pool_kernel(const int* __restrict__ x, const float* __restrict__ V,
                 const float* __restrict__ beta, float* __restrict__ pooled) {
    const int b = blockIdx.x, tid = threadIdx.x;
    const int* xb = x + b * LEAM_L;
    const float* bb = beta + b * LEAM_L;
    float a0 = 0.f, a1 = 0.f;
    for (int l = 0; l < LEAM_L; ++l) {
        const float w = bb[l];
        const float* row = V + (long long)xb[l] * LEAM_P;
        a0 += w * row[tid];
        a1 += w * row[tid + 256];
    }
    pooled[b * LEAM_P + tid]       = a0 * (1.f / (float)LEAM_L);
    pooled[b * LEAM_P + tid + 256] = a1 * (1.f / (float)LEAM_L);
}

// ---------------------------------------------------------------- kernel 4
// out[b,k] = pooled[b,:] . f2_w[k,:] + f2_b[k]
__global__ void __launch_bounds__(64)
leam_head_kernel(const float* __restrict__ pooled, const float* __restrict__ f2_w,
                 const float* __restrict__ f2_b, float* __restrict__ out) {
    const int b = blockIdx.x, k = threadIdx.x;
    const float* p = pooled + b * LEAM_P;
    const float* w = f2_w + k * LEAM_P;
    float acc = 0.f;
    for (int i = 0; i < LEAM_P; ++i) acc += p[i] * w[i];
    out[b * LEAM_K + k] = acc + f2_b[k];
}

// ---------------------------------------------------------------- launcher
extern "C" void kernel_launch(void* const* d_in, const int* in_sizes, int n_in,
                              void* d_out, int out_size, void* d_ws, size_t ws_size,
                              hipStream_t stream) {
    (void)in_sizes; (void)n_in; (void)out_size; (void)ws_size;
    const int*   x   = (const int*)  d_in[0];
    const float* V   = (const float*)d_in[1];
    const float* C   = (const float*)d_in[2];
    const float* f1w = (const float*)d_in[3];
    const float* f1b = (const float*)d_in[4];
    const float* f2w = (const float*)d_in[5];
    const float* f2b = (const float*)d_in[6];
    float* out = (float*)d_out;

    char* ws = (char*)d_ws;
    _Float16* Cn   = (_Float16*)ws;                                  // 64 KB
    float*    G    = (float*)(ws + 64 * 1024);                       // 16 MB
    float*    beta = (float*)(ws + 64 * 1024 +
                              (size_t)LEAM_B * LEAM_K * LEAM_L * 4); // 256 KB
    float*    pooled = beta + LEAM_B * LEAM_L;                       // 64 KB

    leam_cnorm_kernel<<<LEAM_K, 32, 0, stream>>>(C, Cn);

    const size_t smem = (size_t)2 * 64 * ASTR * sizeof(_Float16);    // 133,120 B
    leam_sim_kernel<<<LEAM_B * (LEAM_L / 64), 64, smem, stream>>>(x, V, Cn, G);

    leam_score_kernel<<<LEAM_B, 256, 0, stream>>>(G, f1w, f1b, beta);
    leam_pool_kernel<<<LEAM_B, 256, 0, stream>>>(x, V, beta, pooled);
    leam_head_kernel<<<LEAM_B, 64, 0, stream>>>(pooled, f2w, f2b, out);
}
